// TransformerUnit_58858231825045
// MI455X (gfx1250) — matrix operand order repaired and checked by reading, not compiled, not run on hardware
//
#include <hip/hip_runtime.h>
#include <hip/hip_bf16.h>
#include <math.h>

typedef __bf16 bf16_t;
typedef __attribute__((ext_vector_type(16))) __bf16 v16bf;
typedef __attribute__((ext_vector_type(8)))  __bf16 v8bf;
typedef __attribute__((ext_vector_type(8)))  float  v8f;

#define BB 4
#define SS 4096
#define DD 384
#define HH 6
#define HD 64
#define FF 1536
#define MROWS (BB*SS)   /* 16384 */

static __device__ __forceinline__ bf16_t f2bf(float f){ return (bf16_t)f; }

static __device__ __forceinline__ v8f wmma_bf16(v16bf a, v16bf b, v8f c){
  return __builtin_amdgcn_wmma_f32_16x16x32_bf16(false, a, false, b, (short)0, c, false, false);
}

// A-fragment (16x32 bf16): lane holds row (lane&15); K chunk [16*half,+16) contiguous
static __device__ __forceinline__ v16bf load_a_frag(const bf16_t* rowptr, int kk, int half){
  v16bf a;
  *(v8bf*)&a         = *(const v8bf*)(rowptr + kk +     half*16);
  *(((v8bf*)&a) + 1) = *(const v8bf*)(rowptr + kk + 8 + half*16);
  return a;
}

// one lane of an async 16B global->LDS copy (ASYNCcnt-tracked DMA into LDS)
static __device__ __forceinline__ void async_copy_b128(unsigned lds_off, const bf16_t* gsrc){
  asm volatile("global_load_async_to_lds_b128 %0, %1, off"
               :: "v"(lds_off), "v"((unsigned long long)(size_t)gsrc)
               : "memory");
}
static __device__ __forceinline__ void wait_async0(){
  asm volatile("s_wait_asynccnt 0x0" ::: "memory");
}

// ---------------------------------------------------------------------------
// Block-cooperative GEMM: C(16x64 per wave) = A[M,K] x Bt[N,K]^T.
// All 8 waves of the block share one 64-column B panel (uniform n0); B k-chunks
// are double-buffered in LDS via GLOBAL_LOAD_ASYNC_TO_LDS_B128. The loop is
// manually unrolled x2 over the two buffers and kept branch-free (clamped
// next-stage index) so accumulators stay pinned in VGPRs (no phi copies).
// ---------------------------------------------------------------------------
static __device__ __forceinline__ void gemm_block_16x64(
    const bf16_t* __restrict__ A, int lda, int m0,
    const bf16_t* __restrict__ Bt, int ldb, int n0,
    int K, v8f acc[4])
{
  __shared__ bf16_t Bs[2][64][32];       // 8 KB, double-buffered B panel chunk
  const int tid  = threadIdx.x;
  const int lane = tid & 31, l15 = lane & 15, half = lane >> 4;

  const int col = tid >> 2, seg = tid & 3;        // 256 threads -> 64x32 chunk
  const bf16_t* gcol = Bt + (size_t)(n0 + col) * ldb + seg*8;
  const bf16_t* arow = A + (size_t)(m0 + l15) * lda;
  const int nsteps = K >> 5;                      // always even here (12 / 48)

  // stage chunk 0 into buffer 0
  async_copy_b128((unsigned)(size_t)&Bs[0][col][seg*8], gcol);

  #pragma unroll 1
  for (int i = 0; i < nsteps; i += 2){
    // ---- chunk i (buffer 0) ----
    wait_async0();               // my wave's chunk-i copies have landed
    __syncthreads();             // block-wide visibility; other buffer is free
    {
      int nx = (i + 1 < nsteps) ? (i + 1) : (nsteps - 1);   // branchless clamp
      async_copy_b128((unsigned)(size_t)&Bs[1][col][seg*8], gcol + (size_t)nx*32);
      v16bf a = load_a_frag(arow, i*32, half);
      v16bf bfr[4];
      #pragma unroll
      for (int n = 0; n < 4; ++n)
        bfr[n] = *(const v16bf*)&Bs[0][n*16 + l15][half*16];
      #pragma unroll
      for (int n = 0; n < 4; ++n)
        acc[n] = wmma_bf16(a, bfr[n], acc[n]);
    }
    // ---- chunk i+1 (buffer 1) ----
    wait_async0();
    __syncthreads();
    {
      int nx = (i + 2 < nsteps) ? (i + 2) : (nsteps - 1);   // branchless clamp
      async_copy_b128((unsigned)(size_t)&Bs[0][col][seg*8], gcol + (size_t)nx*32);
      v16bf a = load_a_frag(arow, (i+1)*32, half);
      v16bf bfr[4];
      #pragma unroll
      for (int n = 0; n < 4; ++n)
        bfr[n] = *(const v16bf*)&Bs[1][n*16 + l15][half*16];
      #pragma unroll
      for (int n = 0; n < 4; ++n)
        acc[n] = wmma_bf16(a, bfr[n], acc[n]);
    }
  }
}

// ---------------- weight convert + transpose: W[b][K][N] fp32 -> Wt[b][N][K] bf16 ----------------
__global__ void k_cvt_t(const float* __restrict__ W, bf16_t* __restrict__ Wt,
                        int K, int N, long total){
  long i = (long)blockIdx.x*blockDim.x + threadIdx.x;
  if (i >= total) return;
  int n = (int)(i % N); long r = i / N; int k = (int)(r % K); int b = (int)(r / K);
  Wt[((size_t)b*N + n)*K + k] = f2bf(W[((size_t)b*K + k)*N + n]);
}

// ---------------- LayerNorm (wave per row, D=384 -> 12 elems/lane) ----------------
__global__ void k_ln(const float* __restrict__ x, const float* __restrict__ g,
                     const float* __restrict__ be, bf16_t* __restrict__ out){
  int wid = threadIdx.x >> 5, lane = threadIdx.x & 31;
  int row = blockIdx.x*8 + wid;
  const float* xr = x + (size_t)row*DD;
  float v[12]; float s = 0.f;
  #pragma unroll
  for (int i=0;i<12;++i){ v[i] = xr[lane + i*32]; s += v[i]; }
  #pragma unroll
  for (int o=1;o<32;o<<=1) s += __shfl_xor(s, o, 32);
  float mu = s * (1.f/DD);
  float q = 0.f;
  #pragma unroll
  for (int i=0;i<12;++i){ float d = v[i]-mu; q += d*d; }
  #pragma unroll
  for (int o=1;o<32;o<<=1) q += __shfl_xor(q, o, 32);
  float inv = rsqrtf(q*(1.f/DD) + 1e-5f);
  bf16_t* orow = out + (size_t)row*DD;
  #pragma unroll
  for (int i=0;i<12;++i){ int c = lane + i*32; orow[c] = f2bf((v[i]-mu)*inv*g[c] + be[c]); }
}

// ---------------- QKV projection: h1[16384,384] x Wqkv_t[3*6*64,384] ----------------
__global__ void k_qkv(const bf16_t* __restrict__ h1, const bf16_t* __restrict__ wqkv_t,
                      bf16_t* __restrict__ qo, bf16_t* __restrict__ ko, bf16_t* __restrict__ vto){
  int lane = threadIdx.x & 31, l15 = lane & 15, half = lane >> 4;
  int w = blockIdx.x*8 + (threadIdx.x>>5);
  int mt = w & 1023; int g = w >> 10;        // 1024 m-tiles x 18 n-groups
  int t = g/HH, h = g%HH;
  v8f acc[4] = {};
  const bf16_t* Bt = wqkv_t + (size_t)((t*HH + h)*HD) * DD;
  gemm_block_16x64(h1, DD, mt*16, Bt, DD, 0, DD, acc);
  #pragma unroll
  for (int r=0;r<8;++r){
    int m = mt*16 + r + 8*half; int b = m >> 12; int s = m & 4095;
    #pragma unroll
    for (int n=0;n<4;++n){
      int col = n*16 + l15; bf16_t v = f2bf(acc[n][r]);
      if (t == 0)      qo [((size_t)(b*HH+h)*SS + s)*HD + col] = v;
      else if (t == 1) ko [((size_t)(b*HH+h)*SS + s)*HD + col] = v;
      else             vto[((size_t)(b*HH+h)*HD + col)*SS + s] = v;  // V transposed
    }
  }
}

// ---------------- Flash attention: wave = 16 query rows, 32-key chunks ----------------
static __device__ __forceinline__ void load_k_frags(const bf16_t* kp, int kc,
                                                    int l15, int half, v16bf bk[2][2]){
  #pragma unroll
  for (int n2=0;n2<2;++n2){
    const bf16_t* kcol = kp + (size_t)(kc + n2*16 + l15)*HD;
    bk[n2][0] = *(const v16bf*)(kcol +      half*16);
    bk[n2][1] = *(const v16bf*)(kcol + 32 + half*16);
  }
}

__global__ void k_attn(const bf16_t* __restrict__ q, const bf16_t* __restrict__ k,
                       const bf16_t* __restrict__ vt, bf16_t* __restrict__ attn){
  __shared__ bf16_t Pb[8][16*32];
  int wid = threadIdx.x >> 5, lane = threadIdx.x & 31;
  int l15 = lane & 15, half = lane >> 4;
  int w = blockIdx.x*8 + wid;
  int qt = w & 255; int bh = w >> 8;         // 256 q-tiles per (b,h)
  int q0 = qt*16;
  const bf16_t* qp = q  + (size_t)bh*SS*HD;
  const bf16_t* kp = k  + (size_t)bh*SS*HD;
  const bf16_t* vp = vt + (size_t)bh*HD*SS;
  int b = bh/HH, h = bh%HH;
  const bf16_t* qrow = qp + (size_t)(q0 + l15)*HD;
  v16bf aq0 = load_a_frag(qrow, 0,  half);
  v16bf aq1 = load_a_frag(qrow, 32, half);
  v8f acc[4] = {};
  float mstat[8], lstat[8];
  #pragma unroll
  for (int r=0;r<8;++r){ mstat[r] = -1e30f; lstat[r] = 0.f; }
  bf16_t* P = Pb[wid];

  v16bf bk[2][2];
  load_k_frags(kp, 0, l15, half, bk);        // prefetch first K chunk

  for (int kc = 0; kc < q0 + 16; kc += 32){
    // V fragments for this chunk issued early (consumed after softmax)
    v16bf bv[4];
    #pragma unroll
    for (int n=0;n<4;++n)
      bv[n] = *(const v16bf*)(vp + (size_t)(n*16 + l15)*SS + kc + half*16);

    // scores: Q(16x64) x K^T(64x32) -> two 16x16 C tiles
    v8f sc[2];
    #pragma unroll
    for (int n2=0;n2<2;++n2){
      v8f c = {};
      c = wmma_bf16(aq0, bk[n2][0], c);
      c = wmma_bf16(aq1, bk[n2][1], c);
      sc[n2] = c;
    }
    // prefetch next chunk's K fragments; softmax VALU below hides the latency
    if (kc + 32 < q0 + 16) load_k_frags(kp, kc + 32, l15, half, bk);

    bool need_mask = (kc + 31) > q0;
    #pragma unroll
    for (int r=0;r<8;++r){
      int row = q0 + r + 8*half;
      #pragma unroll
      for (int n2=0;n2<2;++n2){
        float s = sc[n2][r] * 0.125f;          // hd^-0.5
        int key = kc + n2*16 + l15;
        if (need_mask && key > row) s = -1e30f;
        sc[n2][r] = s;
      }
    }
    // online softmax (row stats per C-register / half-wave; width-16 reductions)
    #pragma unroll
    for (int r=0;r<8;++r){
      float t = fmaxf(sc[0][r], sc[1][r]);
      #pragma unroll
      for (int o=1;o<16;o<<=1) t = fmaxf(t, __shfl_xor(t, o, 16));
      float mnew  = fmaxf(mstat[r], t);
      float scale = __expf(mstat[r] - mnew);
      float p0 = __expf(sc[0][r] - mnew);
      float p1 = __expf(sc[1][r] - mnew);
      float rs = p0 + p1;
      #pragma unroll
      for (int o=1;o<16;o<<=1) rs += __shfl_xor(rs, o, 16);
      lstat[r] = lstat[r]*scale + rs;
      mstat[r] = mnew;
      #pragma unroll
      for (int n=0;n<4;++n) acc[n][r] *= scale;
      P[(r + 8*half)*32 +      l15] = f2bf(p0);   // C-layout -> LDS [row][key]
      P[(r + 8*half)*32 + 16 + l15] = f2bf(p1);
    }
    asm volatile("s_wait_dscnt 0x0" ::: "memory");  // same-wave LDS is in-order
    v16bf ap;                                       // reload P in A-fragment layout
    *(v8bf*)&ap         = *(const v8bf*)(P + l15*32 +     half*16);
    *(((v8bf*)&ap) + 1) = *(const v8bf*)(P + l15*32 + 8 + half*16);
    #pragma unroll
    for (int n=0;n<4;++n)
      acc[n] = wmma_bf16(ap, bv[n], acc[n]);
  }
  #pragma unroll
  for (int r=0;r<8;++r){
    float inv = 1.f / lstat[r];
    int m = b*SS + q0 + r + 8*half;
    #pragma unroll
    for (int n=0;n<4;++n)
      attn[(size_t)m*DD + h*HD + n*16 + l15] = f2bf(acc[n][r]*inv);
  }
}

// ---------------- O-proj: x1 = x + attn x Wo + bo (fp32 out) ----------------
__global__ void k_oproj(const bf16_t* __restrict__ attn, const bf16_t* __restrict__ wo_t,
                        const float* __restrict__ x, const float* __restrict__ bo,
                        float* __restrict__ x1){
  int lane = threadIdx.x & 31, l15 = lane & 15, half = lane >> 4;
  int w = blockIdx.x*8 + (threadIdx.x>>5);
  int mt = w & 1023; int g = w >> 10;   // 6 n-groups
  v8f acc[4] = {};
  gemm_block_16x64(attn, DD, mt*16, wo_t, DD, g*64, DD, acc);
  #pragma unroll
  for (int r=0;r<8;++r){
    int m = mt*16 + r + 8*half;
    #pragma unroll
    for (int n=0;n<4;++n){
      int col = g*64 + n*16 + l15;
      x1[(size_t)m*DD + col] = x[(size_t)m*DD + col] + bo[col] + acc[n][r];
    }
  }
}

// ---------------- FFN1: u = relu(h2 x W1 + b1), bf16 out ----------------
__global__ void k_ffn1(const bf16_t* __restrict__ h2, const bf16_t* __restrict__ w1_t,
                       const float* __restrict__ b1, bf16_t* __restrict__ u){
  int lane = threadIdx.x & 31, l15 = lane & 15, half = lane >> 4;
  int w = blockIdx.x*8 + (threadIdx.x>>5);
  int mt = w & 1023; int g = w >> 10;   // 24 n-groups
  v8f acc[4] = {};
  gemm_block_16x64(h2, DD, mt*16, w1_t, DD, g*64, DD, acc);
  #pragma unroll
  for (int r=0;r<8;++r){
    int m = mt*16 + r + 8*half;
    #pragma unroll
    for (int n=0;n<4;++n){
      int col = g*64 + n*16 + l15;
      u[(size_t)m*FF + col] = f2bf(fmaxf(acc[n][r] + b1[col], 0.f));
    }
  }
}

// ---------------- FFN2: out = x1 + u x W2 + b2 (fp32 out) ----------------
__global__ void k_ffn2(const bf16_t* __restrict__ u, const bf16_t* __restrict__ w2_t,
                       const float* __restrict__ b2, const float* __restrict__ x1,
                       float* __restrict__ out){
  int lane = threadIdx.x & 31, l15 = lane & 15, half = lane >> 4;
  int w = blockIdx.x*8 + (threadIdx.x>>5);
  int mt = w & 1023; int g = w >> 10;   // 6 n-groups
  v8f acc[4] = {};
  gemm_block_16x64(u, FF, mt*16, w2_t, FF, g*64, FF, acc);
  #pragma unroll
  for (int r=0;r<8;++r){
    int m = mt*16 + r + 8*half;
    #pragma unroll
    for (int n=0;n<4;++n){
      int col = g*64 + n*16 + l15;
      out[(size_t)m*DD + col] = x1[(size_t)m*DD + col] + b2[col] + acc[n][r];
    }
  }
}

extern "C" void kernel_launch(void* const* d_in, const int* in_sizes, int n_in,
                              void* d_out, int out_size, void* d_ws, size_t ws_size,
                              hipStream_t stream) {
  (void)in_sizes; (void)n_in; (void)out_size; (void)ws_size;
  const float* x   = (const float*)d_in[0];
  const float* Wq  = (const float*)d_in[1];
  const float* Wk  = (const float*)d_in[2];
  const float* Wv  = (const float*)d_in[3];
  const float* Wo  = (const float*)d_in[4];
  const float* bo  = (const float*)d_in[5];
  const float* W1  = (const float*)d_in[6];
  const float* b1  = (const float*)d_in[7];
  const float* W2  = (const float*)d_in[8];
  const float* b2  = (const float*)d_in[9];
  const float* g1  = (const float*)d_in[10];
  const float* be1 = (const float*)d_in[11];
  const float* g2  = (const float*)d_in[12];
  const float* be2 = (const float*)d_in[13];
  float* out = (float*)d_out;

  char* ws = (char*)d_ws;
  bf16_t* wqkv_t = (bf16_t*)(ws + 0);              // [3][6][64][384]  884736 B
  bf16_t* wo_t   = (bf16_t*)(ws + 884736);         // [384][384]       294912 B
  bf16_t* w1_t   = (bf16_t*)(ws + 1179648);        // [1536][384]     1179648 B
  bf16_t* w2_t   = (bf16_t*)(ws + 2359296);        // [384][1536]     1179648 B
  bf16_t* h1     = (bf16_t*)(ws + 3538944);        // [16384][384]   12582912 B
  bf16_t* qb     = (bf16_t*)(ws + 16121856);       // [4][6][4096][64]
  bf16_t* kb     = (bf16_t*)(ws + 28704768);
  bf16_t* vtb    = (bf16_t*)(ws + 41287680);       // [4][6][64][4096]
  bf16_t* attn   = (bf16_t*)(ws + 53870592);       // [16384][384]
  float*  x1     = (float*) (ws + 66453504);       // [16384][384] fp32
  bf16_t* h2     = (bf16_t*)(ws + 91619328);       // [16384][384]
  bf16_t* ub     = (bf16_t*)(ws + 104202240);      // [16384][1536]

  // 1) weights -> bf16, transposed [N][K]
  {
    long t1 = (long)HH*DD*HD;
    k_cvt_t<<<(t1+255)/256, 256, 0, stream>>>(Wq, wqkv_t,                      DD, HD, t1);
    k_cvt_t<<<(t1+255)/256, 256, 0, stream>>>(Wk, wqkv_t + (size_t)HH*HD*DD,   DD, HD, t1);
    k_cvt_t<<<(t1+255)/256, 256, 0, stream>>>(Wv, wqkv_t + (size_t)2*HH*HD*DD, DD, HD, t1);
    long t2 = (long)DD*DD;
    k_cvt_t<<<(t2+255)/256, 256, 0, stream>>>(Wo, wo_t, DD, DD, t2);
    long t3 = (long)DD*FF;
    k_cvt_t<<<(t3+255)/256, 256, 0, stream>>>(W1, w1_t, DD, FF, t3);
    k_cvt_t<<<(t3+255)/256, 256, 0, stream>>>(W2, w2_t, FF, DD, t3);
  }
  // 2) LN1
  k_ln<<<MROWS/8, 256, 0, stream>>>(x, g1, be1, h1);
  // 3) QKV projection (1024 m-tiles x 18 n-groups, 8 waves/block)
  k_qkv<<<(1024*18)/8, 256, 0, stream>>>(h1, wqkv_t, qb, kb, vtb);
  // 4) causal flash attention (24 bh x 256 q-tiles)
  k_attn<<<(BB*HH*256)/8, 256, 0, stream>>>(qb, kb, vtb, attn);
  // 5) O-proj + residual
  k_oproj<<<(1024*6)/8, 256, 0, stream>>>(attn, wo_t, x, bo, x1);
  // 6) LN2
  k_ln<<<MROWS/8, 256, 0, stream>>>(x1, g2, be2, h2);
  // 7) FFN1 + ReLU
  k_ffn1<<<(1024*24)/8, 256, 0, stream>>>(h2, w1_t, b1, ub);
  // 8) FFN2 + residual -> out
  k_ffn2<<<(1024*6)/8, 256, 0, stream>>>(ub, w2_t, b2, x1, out);
}
